// RouteEncoder_41154376630820
// MI455X (gfx1250) — matrix operand order repaired and checked by reading, not compiled, not run on hardware
//
#include <hip/hip_runtime.h>
#include <hip/hip_bf16.h>
#include <stdint.h>

// ---------------------------------------------------------------------------
// RouteEncoder fused kernel for MI455X (gfx1250, wave32, WMMA 16x16x32 f16)
//
// Compute-bound workload (~18 GFLOP vs ~25 MB HBM): use f16 WMMA (K=32/instr,
// 8x the MACs of the f32 16x16x4 path) with f32 accumulation. One workgroup
// per batch row; all intermediates live in LDS (~93 KB). Weights are packed
// once per launch into *fragment-swizzled* f16 so every wave loads its B
// fragment as two coalesced global_load_b128.
// ---------------------------------------------------------------------------

typedef __attribute__((ext_vector_type(16))) _Float16 v16h;
typedef __attribute__((ext_vector_type(8)))  float    v8f;

#define B_    4096
#define PV_   250
#define PVP   256      // PV padded (K of token GEMM)
#define CH_   64
#define TOK_  33
#define TOKP  64       // token dim padded
#define HID_  256
#define NTL_  9
#define NLT_  20
#define EPS_  1e-5f

__device__ __forceinline__ float gelu_exact(float x) {
  return 0.5f * x * (1.0f + erff(x * 0.70710678118654752f));
}

union H2U { uint32_t u; _Float16 h[2]; };

// A fragment (f16, 16x32 tile) from row-major [M][lda] in LDS.
// ISA 7.12.2: M = lane%16; lanes 0-15 K-pairs {0..7,16..23}, lanes 16-31 +8.
// K pairs are contiguous -> 8x ds_load_b32 per fragment.
__device__ __forceinline__ v16h load_frag_a(const _Float16* src, int lda,
                                            int m0, int k0, int lane) {
  v16h a;
  const _Float16* row = src + (size_t)(m0 + (lane & 15)) * lda + k0 + ((lane >> 4) << 3);
#pragma unroll
  for (int v = 0; v < 4; ++v) {
    H2U lo, hi;
    lo.u = *(const uint32_t*)(row + 2 * v);        // K = 2v, 2v+1 (+8g)
    hi.u = *(const uint32_t*)(row + 16 + 2 * v);   // K = 16+2v   (+8g)
    a[2 * v]         = lo.h[0];
    a[2 * v + 1]     = lo.h[1];
    a[8 + 2 * v]     = hi.h[0];
    a[8 + 2 * v + 1] = hi.h[1];
  }
  return a;
}

// One 16x16 output tile against a fragment-swizzled weight matrix.
// Swizzled layout: frag (nt,ks) -> 32 lanes x 16 halves, lane-contiguous,
// so the whole B fragment is one aligned 32-byte load (2x global_load_b128).
template <int KSTEPS>
__device__ __forceinline__ v8f gemm_tile_sw(const _Float16* A, int lda, int m0,
                                            const _Float16* Wsw, int nt,
                                            int lane) {
  v8f c = {};
#pragma unroll
  for (int ks = 0; ks < KSTEPS; ++ks) {
    v16h a = load_frag_a(A, lda, m0, ks * 32, lane);
    v16h b = *((const v16h*)Wsw + ((size_t)(nt * KSTEPS + ks) * 32 + lane));
    c = __builtin_amdgcn_wmma_f32_16x16x32_f16(false, a, false, b,
                                               (short)0, c, false, false);
  }
  return c;
}

// ---------------------------------------------------------------------------
// Weight packer: fp32 (sr x sc) -> f16, fragment-swizzled, zero padded.
// dst index = ((nt*KS + ks)*32 + lane)*16 + e ; e -> (v,h):
//   k = ks*32 + 8*(lane/16) + (e&7) + 16*(e/8) ; n = nt*16 + lane%16
// ---------------------------------------------------------------------------
__global__ void re_pack_w16_sw(const float* __restrict__ src,
                               _Float16* __restrict__ dst,
                               int sr, int sc, int KS, int NT) {
  int i = blockIdx.x * blockDim.x + threadIdx.x;
  int total = KS * NT * 32 * 16;
  if (i >= total) return;
  int e    = i & 15;
  int lane = (i >> 4) & 31;
  int f    = i >> 9;
  int ks   = f % KS;
  int nt   = f / KS;
  int k = ks * 32 + ((lane >> 4) << 3) + (e & 7) + ((e >> 3) << 4);
  int n = nt * 16 + (lane & 15);
  float val = (k < sr && n < sc) ? src[k * sc + n] : 0.0f;
  dst[i] = (_Float16)val;
}

// ---------------------------------------------------------------------------
// Fused per-batch-row kernel
// ---------------------------------------------------------------------------
struct REParams {
  const float* x;
  const float* chW1;  const float* chb1;  const float* chb2;
  const float* tokb1; const float* tokb2;
  const float* tl_emb; const float* lt_emb;
  const float* n1g; const float* n1b;
  const float* mtb1; const float* mtb2;
  const float* n2g; const float* n2b;
  const float* mcb1; const float* mcb2;
  const float* ng;  const float* nb;
  const float* outW1; const float* outb1;
  const float* outW2; const float* outb2;
  const _Float16* wChW2;   // 64x64   swizzled (KS=2, NT=4)
  const _Float16* wTokW1;  // 256x64  swizzled (KS=8, NT=4)
  const _Float16* wTokW2;  // 64x64   swizzled
  const _Float16* wMTW1;   // 64x64   swizzled
  const _Float16* wMTW2;   // 64x64   swizzled
  const _Float16* wMCW1;   // 64x64   swizzled
  const _Float16* wMCW2;   // 64x64   swizzled
  float* out;
};

__global__ __launch_bounds__(256) void route_encoder_fused(REParams p) {
  const int b    = blockIdx.x;
  const int tid  = threadIdx.x;
  const int lane = tid & 31;
  const int wave = tid >> 5;

  __shared__ _Float16 sA [PVP * CH_];   // 32 KB  point features (M=256, K=64)
  __shared__ _Float16 sF0[CH_ * PVP];   // 32 KB  f0^T           (M=64,  K=256)
  __shared__ _Float16 sX [64 * 64];     //  8 KB  scratch A
  __shared__ _Float16 sY [64 * 64];     //  8 KB  scratch A
  __shared__ float    sf1[TOK_ * CH_];  //  8.25 KB  f (tok, ch)
  __shared__ float    sm [CH_];
  __shared__ float    sml[CH_];
  __shared__ float    shid[HID_];
  __shared__ short    sTL[PVP];
  __shared__ short    sLT[PVP];
  __shared__ int      svalid;

  if (tid == 0) svalid = 0;
  // warm caches with the big packed weight (global_prefetch_b8)
  __builtin_prefetch(p.wTokW1 + (tid << 6), 0, 0);
  __syncthreads();

  // ---- Stage 1: featurize + dense(4->64) + gelu; gather emb indices -------
  {
    const int pnt = tid;  // one point per thread (256 >= 250)
    float c0 = 0.f, c1 = 0.f, g2 = 0.f, g3 = 0.f;
    if (pnt < PV_) {
      const float* xr = p.x + ((size_t)b * PV_ + pnt) * 5;
      c0 = xr[0]; c1 = xr[1];
      float hd = xr[2];
      g2 = cosf(hd); g3 = sinf(hd);
      int nz = (c0 != 0.f) + (c1 != 0.f) + (g2 != 0.f) + (g3 != 0.f);
      if (nz != 0) atomicOr(&svalid, 1);
      int tlv = (int)xr[3];
      tlv = tlv < 0 ? 0 : (tlv > NTL_ - 1 ? NTL_ - 1 : tlv);
      int ltv = (int)xr[4];
      ltv = ltv < 0 ? 0 : (ltv > NLT_ - 1 ? NLT_ - 1 : ltv);
      sTL[pnt] = (short)tlv;
      sLT[pnt] = (short)ltv;
#pragma unroll 4
      for (int j = 0; j < CH_; j += 2) {
        float s0 = c0 * p.chW1[0 * CH_ + j] + c1 * p.chW1[1 * CH_ + j] +
                   g2 * p.chW1[2 * CH_ + j] + g3 * p.chW1[3 * CH_ + j] +
                   p.chb1[j];
        float s1 = c0 * p.chW1[0 * CH_ + j + 1] + c1 * p.chW1[1 * CH_ + j + 1] +
                   g2 * p.chW1[2 * CH_ + j + 1] + g3 * p.chW1[3 * CH_ + j + 1] +
                   p.chb1[j + 1];
        H2U u;
        u.h[0] = (_Float16)gelu_exact(s0);
        u.h[1] = (_Float16)gelu_exact(s1);
        *(uint32_t*)&sA[pnt * CH_ + j] = u.u;   // paired ds_store_b32
      }
    } else {
      sTL[pnt] = 0;
      sLT[pnt] = 0;
      uint32_t* row32 = (uint32_t*)&sA[pnt * CH_];
#pragma unroll
      for (int j = 0; j < CH_ / 2; ++j) row32[j] = 0u;
    }
  }
  __syncthreads();

  // ---- GEMM0: (256x64) @ chW2(64x64) + b2 + emb  -> sF0 transposed --------
  for (int t = wave; t < 64; t += 8) {
    int mt = t >> 2, nt = t & 3;
    v8f c = gemm_tile_sw<2>(sA, CH_, mt * 16, p.wChW2, nt, lane);
    int col   = nt * 16 + (lane & 15);
    int rbase = mt * 16 + ((lane >> 4) << 3);
    float bias = p.chb2[col];
#pragma unroll
    for (int r = 0; r < 8; ++r) {
      int row = rbase + r;  // point index (pad rows have idx 0 -> safe loads)
      float e = p.tl_emb[(int)sTL[row] * CH_ + col] +
                p.lt_emb[(int)sLT[row] * CH_ + col];
      float val = c[r] + bias + e;
      sF0[col * PVP + row] = (row < PV_) ? (_Float16)val : (_Float16)0.f;
    }
  }
  __syncthreads();

  // ---- GEMM1: sF0(64x256) @ tokW1(256x64) -> gelu -> sX [ch][tokp] --------
  for (int t = wave; t < 16; t += 8) {
    int mt = t >> 2, nt = t & 3;
    v8f c = gemm_tile_sw<8>(sF0, PVP, mt * 16, p.wTokW1, nt, lane);
    int col   = nt * 16 + (lane & 15);
    int rbase = mt * 16 + ((lane >> 4) << 3);
    float bias = (col < TOK_) ? p.tokb1[col] : 0.f;
#pragma unroll
    for (int r = 0; r < 8; ++r)
      sX[(rbase + r) * TOKP + col] = (_Float16)gelu_exact(c[r] + bias);
  }
  __syncthreads();

  // ---- GEMM2: sX(64x64) @ tokW2 + b2 -> sf1[tok][ch] ----------------------
  for (int t = wave; t < 16; t += 8) {
    int mt = t >> 2, nt = t & 3;
    v8f c = gemm_tile_sw<2>(sX, TOKP, mt * 16, p.wTokW2, nt, lane);
    int col = nt * 16 + (lane & 15);
    if (col < TOK_) {
      int rbase = mt * 16 + ((lane >> 4) << 3);
      float bias = p.tokb2[col];
#pragma unroll
      for (int r = 0; r < 8; ++r)
        sf1[col * CH_ + (rbase + r)] = c[r] + bias;
    }
  }
  __syncthreads();

  // ---- LN1 over channels, transposed into sX [ch][tokp] (padded) ----------
  {
    uint32_t* sX32 = (uint32_t*)sX;
    for (int i = tid; i < 64 * 64 / 2; i += 256) sX32[i] = 0u;
  }
  __syncthreads();
  if (tid < TOK_) {
    float mu = 0.f;
    for (int j = 0; j < CH_; ++j) mu += sf1[tid * CH_ + j];
    mu *= (1.f / CH_);
    float var = 0.f;
    for (int j = 0; j < CH_; ++j) { float d = sf1[tid * CH_ + j] - mu; var += d * d; }
    var *= (1.f / CH_);
    float rs = rsqrtf(var + EPS_);
    for (int j = 0; j < CH_; ++j)
      sX[j * TOKP + tid] =
          (_Float16)((sf1[tid * CH_ + j] - mu) * rs * p.n1g[j] + p.n1b[j]);
  }
  __syncthreads();

  // ---- GEMM3: sX(64x64) @ MTW1 -> gelu -> sY ------------------------------
  for (int t = wave; t < 16; t += 8) {
    int mt = t >> 2, nt = t & 3;
    v8f c = gemm_tile_sw<2>(sX, TOKP, mt * 16, p.wMTW1, nt, lane);
    int col   = nt * 16 + (lane & 15);
    int rbase = mt * 16 + ((lane >> 4) << 3);
    float bias = (col < TOK_) ? p.mtb1[col] : 0.f;
#pragma unroll
    for (int r = 0; r < 8; ++r)
      sY[(rbase + r) * TOKP + col] = (_Float16)gelu_exact(c[r] + bias);
  }
  __syncthreads();

  // ---- GEMM4: sY @ MTW2 + b2 ; f1 += result^T -----------------------------
  for (int t = wave; t < 16; t += 8) {
    int mt = t >> 2, nt = t & 3;
    v8f c = gemm_tile_sw<2>(sY, TOKP, mt * 16, p.wMTW2, nt, lane);
    int col = nt * 16 + (lane & 15);
    if (col < TOK_) {
      int rbase = mt * 16 + ((lane >> 4) << 3);
      float bias = p.mtb2[col];
#pragma unroll
      for (int r = 0; r < 8; ++r)
        sf1[col * CH_ + (rbase + r)] += c[r] + bias;
    }
  }
  __syncthreads();

  // ---- LN2 over channels -> sX [tokp48][ch] -------------------------------
  {
    uint32_t* sX32 = (uint32_t*)sX;
    for (int i = tid; i < 64 * 64 / 2; i += 256) sX32[i] = 0u;
  }
  __syncthreads();
  if (tid < TOK_) {
    float mu = 0.f;
    for (int j = 0; j < CH_; ++j) mu += sf1[tid * CH_ + j];
    mu *= (1.f / CH_);
    float var = 0.f;
    for (int j = 0; j < CH_; ++j) { float d = sf1[tid * CH_ + j] - mu; var += d * d; }
    var *= (1.f / CH_);
    float rs = rsqrtf(var + EPS_);
    for (int j = 0; j < CH_; j += 2) {
      H2U u;
      u.h[0] = (_Float16)((sf1[tid * CH_ + j] - mu) * rs * p.n2g[j] + p.n2b[j]);
      u.h[1] = (_Float16)((sf1[tid * CH_ + j + 1] - mu) * rs * p.n2g[j + 1] + p.n2b[j + 1]);
      *(uint32_t*)&sX[tid * CH_ + j] = u.u;
    }
  }
  __syncthreads();

  // ---- GEMM5: (48x64) @ MCW1 -> gelu -> sY --------------------------------
  for (int t = wave; t < 12; t += 8) {
    int mt = t >> 2, nt = t & 3;
    v8f c = gemm_tile_sw<2>(sX, CH_, mt * 16, p.wMCW1, nt, lane);
    int col   = nt * 16 + (lane & 15);
    int rbase = mt * 16 + ((lane >> 4) << 3);
    float bias = p.mcb1[col];
#pragma unroll
    for (int r = 0; r < 8; ++r)
      sY[(rbase + r) * CH_ + col] = (_Float16)gelu_exact(c[r] + bias);
  }
  __syncthreads();

  // ---- GEMM6: (48x64) @ MCW2 + b2 ; f1 += (rows < 33) ---------------------
  for (int t = wave; t < 12; t += 8) {
    int mt = t >> 2, nt = t & 3;
    v8f c = gemm_tile_sw<2>(sY, CH_, mt * 16, p.wMCW2, nt, lane);
    int col   = nt * 16 + (lane & 15);
    int rbase = mt * 16 + ((lane >> 4) << 3);
    float bias = p.mcb2[col];
#pragma unroll
    for (int r = 0; r < 8; ++r) {
      int row = rbase + r;
      if (row < TOK_) sf1[row * CH_ + col] += c[r] + bias;
    }
  }
  __syncthreads();

  // ---- mean over tokens ---------------------------------------------------
  if (tid < CH_) {
    float s = 0.f;
    for (int t2 = 0; t2 < TOK_; ++t2) s += sf1[t2 * CH_ + tid];
    sm[tid] = s * (1.f / TOK_);
  }
  __syncthreads();

  // ---- LN3 (redundant per-thread mu/var over 64 values) -------------------
  {
    float mu = 0.f;
    for (int k = 0; k < CH_; ++k) mu += sm[k];
    mu *= (1.f / CH_);
    float var = 0.f;
    for (int k = 0; k < CH_; ++k) { float d = sm[k] - mu; var += d * d; }
    var *= (1.f / CH_);
    float rs = rsqrtf(var + EPS_);
    if (tid < CH_) sml[tid] = (sm[tid] - mu) * rs * p.ng[tid] + p.nb[tid];
  }
  __syncthreads();

  // ---- out MLP (64 -> 256 -> 256), VALU, coalesced across threads ---------
  {
    float acc = p.outb1[tid];
    for (int k = 0; k < CH_; ++k) acc += sml[k] * p.outW1[k * HID_ + tid];
    shid[tid] = gelu_exact(acc);
  }
  __syncthreads();
  {
    float acc = p.outb2[tid];
    for (int k = 0; k < HID_; ++k) acc += shid[k] * p.outW2[k * HID_ + tid];
    p.out[(size_t)b * HID_ + tid] = svalid ? acc : 0.f;
  }
}

// ---------------------------------------------------------------------------
// Host launcher
// ---------------------------------------------------------------------------
extern "C" void kernel_launch(void* const* d_in, const int* in_sizes, int n_in,
                              void* d_out, int out_size, void* d_ws, size_t ws_size,
                              hipStream_t stream) {
  (void)in_sizes; (void)n_in; (void)out_size; (void)ws_size;

  // input order per setup_inputs(): x, then params in insertion order
  const float* x       = (const float*)d_in[0];
  const float* chW1    = (const float*)d_in[1];
  const float* chb1    = (const float*)d_in[2];
  const float* chW2    = (const float*)d_in[3];
  const float* chb2    = (const float*)d_in[4];
  const float* tokW1   = (const float*)d_in[5];
  const float* tokb1   = (const float*)d_in[6];
  const float* tokW2   = (const float*)d_in[7];
  const float* tokb2   = (const float*)d_in[8];
  const float* tl_emb  = (const float*)d_in[9];
  const float* lt_emb  = (const float*)d_in[10];
  const float* n1g     = (const float*)d_in[11];
  const float* n1b     = (const float*)d_in[12];
  const float* mtW1    = (const float*)d_in[13];
  const float* mtb1    = (const float*)d_in[14];
  const float* mtW2    = (const float*)d_in[15];
  const float* mtb2    = (const float*)d_in[16];
  const float* n2g     = (const float*)d_in[17];
  const float* n2b     = (const float*)d_in[18];
  const float* mcW1    = (const float*)d_in[19];
  const float* mcb1    = (const float*)d_in[20];
  const float* mcW2    = (const float*)d_in[21];
  const float* mcb2    = (const float*)d_in[22];
  const float* ng      = (const float*)d_in[23];
  const float* nb      = (const float*)d_in[24];
  const float* outW1   = (const float*)d_in[25];
  const float* outb1   = (const float*)d_in[26];
  const float* outW2   = (const float*)d_in[27];
  const float* outb2   = (const float*)d_in[28];

  // packed, fragment-swizzled f16 weights in workspace
  _Float16* ws      = (_Float16*)d_ws;
  _Float16* wChW2   = ws;
  _Float16* wTokW1  = wChW2  + 64 * 64;
  _Float16* wTokW2  = wTokW1 + 256 * 64;
  _Float16* wMTW1   = wTokW2 + 64 * 64;
  _Float16* wMTW2   = wMTW1  + 64 * 64;
  _Float16* wMCW1   = wMTW2  + 64 * 64;
  _Float16* wMCW2   = wMCW1  + 64 * 64;

  auto pk = [&](const float* src, _Float16* dst, int sr, int sc, int KS, int NT) {
    int n = KS * NT * 32 * 16;
    re_pack_w16_sw<<<(n + 255) / 256, 256, 0, stream>>>(src, dst, sr, sc, KS, NT);
  };
  pk(chW2,  wChW2,  64, 64, 2, 4);    // KS = K/32, NT = N/16
  pk(tokW1, wTokW1, 250, 33, 8, 4);
  pk(tokW2, wTokW2, 33, 33, 2, 4);
  pk(mtW1,  wMTW1,  33, 33, 2, 4);
  pk(mtW2,  wMTW2,  33, 33, 2, 4);
  pk(mcW1,  wMCW1,  64, 64, 2, 4);
  pk(mcW2,  wMCW2,  64, 64, 2, 4);

  REParams p;
  p.x = x;
  p.chW1 = chW1; p.chb1 = chb1; p.chb2 = chb2;
  p.tokb1 = tokb1; p.tokb2 = tokb2;
  p.tl_emb = tl_emb; p.lt_emb = lt_emb;
  p.n1g = n1g; p.n1b = n1b;
  p.mtb1 = mtb1; p.mtb2 = mtb2;
  p.n2g = n2g; p.n2b = n2b;
  p.mcb1 = mcb1; p.mcb2 = mcb2;
  p.ng = ng; p.nb = nb;
  p.outW1 = outW1; p.outb1 = outb1; p.outW2 = outW2; p.outb2 = outb2;
  p.wChW2 = wChW2; p.wTokW1 = wTokW1; p.wTokW2 = wTokW2;
  p.wMTW1 = wMTW1; p.wMTW2 = wMTW2; p.wMCW1 = wMCW1; p.wMCW2 = wMCW2;
  p.out = (float*)d_out;

  route_encoder_fused<<<B_, 256, 0, stream>>>(p);
}
